// Head_30391188587331
// MI455X (gfx1250) — matrix operand order repaired and checked by reading, not compiled, not run on hardware
//
#include <hip/hip_runtime.h>

typedef __attribute__((ext_vector_type(16))) _Float16 v16h;
typedef __attribute__((ext_vector_type(8)))  _Float16 v8h;
typedef __attribute__((ext_vector_type(8)))  float    v8f;

#define B_ 16
#define T_ 2048
#define C_ 68
#define H_ 64

__device__ __forceinline__ v16h cat16(v8h lo, v8h hi) {
    v16h r;
#pragma unroll
    for (int i = 0; i < 8; ++i) { r[i] = lo[i]; r[i + 8] = hi[i]; }
    return r;
}

// ---------------------------------------------------------------------------
// Stage 1: q = xWq, k = xWk, v = xWv  (f32 accumulate, f16 outputs).
// Q is pre-scaled by (1/sqrt(64)) * log2(e): the S^T WMMA output then lands
// directly in the base-2 softmax domain (v_exp_f32 is exp2), so the attention
// loop needs no per-element scale/log2e multiplies at all.
// V is stored transposed per batch: vt[(b*64 + h)*T + t], so the O^T = V^T P^T
// A-fragments become contiguous 16-byte loads per lane.
// ---------------------------------------------------------------------------
__global__ __launch_bounds__(64)
void proj_kernel(const float* __restrict__ x,
                 const float* __restrict__ Wk,
                 const float* __restrict__ Wq,
                 const float* __restrict__ Wv,
                 _Float16* __restrict__ qb,
                 _Float16* __restrict__ kb,
                 _Float16* __restrict__ vt)
{
    __shared__ float xs[C_];
    const int row = blockIdx.x;          // b*T + t
    const int b   = row / T_;
    const int t   = row - b * T_;
    const int h   = threadIdx.x;         // 0..63 = head dim
    for (int c = h; c < C_; c += 64) xs[c] = x[(size_t)row * C_ + c];
    __syncthreads();

    float ks = 0.f, qs = 0.f, vs = 0.f;
#pragma unroll 4
    for (int c = 0; c < C_; ++c) {
        const float xv = xs[c];
        ks = fmaf(xv, Wk[c * H_ + h], ks);
        qs = fmaf(xv, Wq[c * H_ + h], qs);
        vs = fmaf(xv, Wv[c * H_ + h], vs);
    }
    const float qscale = 0.125f * 1.44269504088896340736f;  // 1/sqrt(H) * log2(e)
    qb[(size_t)row * H_ + h] = (_Float16)(qs * qscale);
    kb[(size_t)row * H_ + h] = (_Float16)ks;
    vt[((size_t)b * H_ + h) * T_ + t] = (_Float16)vs;
}

// ---------------------------------------------------------------------------
// Stage 2: flash attention, transposed dataflow, 64-key tiles.
//   S^T = K * Q^T   (keys = M rows, queries = N columns -> one query per lane)
//   O^T = V^T * P^T (head = M rows,  queries = N columns)
// Per-lane softmax stats in the log2 domain; key-dim reduction is an in-lane
// VALU tree plus a single xor-16 shuffle per 64-key tile.
// 1 block = 4 waves; wave w owns a 16-query tile.
// ---------------------------------------------------------------------------
__global__ __launch_bounds__(128)
void attn_kernel(const _Float16* __restrict__ qb,
                 const _Float16* __restrict__ kb,
                 const _Float16* __restrict__ vt,
                 float* __restrict__ out)
{
    __shared__ _Float16 Plds[4][16][64];   // per-wave P^T tile: [query][key]

    const int lane = threadIdx.x & 31;
    const int wv   = threadIdx.x >> 5;
    const int n    = lane & 15;            // query column (N) / M for A-frags
    const int hh   = lane >> 4;            // lane-half select

    const int b     = blockIdx.x >> 5;     // 32 query tiles per batch
    const int qtile = blockIdx.x & 31;
    const int qrow0 = qtile * 64 + wv * 16;
    const size_t qoff = ((size_t)b * T_ + qrow0) * H_;

    // Q^T B-fragments (resident): B[k][n] = Q[qrow0+n][c*32 + k]
    // B layout: elems 0-7 -> k=8hh+e, elems 8-15 -> k=16+8hh+e (matches A).
    v16h Qf[2];
#pragma unroll
    for (int c = 0; c < 2; ++c) {
        const _Float16* p = qb + qoff + (size_t)n * H_ + c * 32 + 8 * hh;
        Qf[c] = cat16(*(const v8h*)p, *(const v8h*)(p + 16));
    }

    v8f O[4];                              // O^T: head 16f+r+8hh x query n
#pragma unroll
    for (int f = 0; f < 4; ++f) O[f] = {};
    float mrun = -1e30f, lrun = 0.f;       // per-lane stats, log2 domain

    const size_t kbat = (size_t)b * T_ * H_;
    const size_t vbat = (size_t)b * H_ * T_;

    for (int key0 = 0; key0 < T_; key0 += 64) {
        if (key0 + 64 < T_)
            __builtin_prefetch(kb + kbat + (size_t)(key0 + 64 + n) * H_, 0, 1);

        // ---- S^T(64x16) = K * Q^T  (already in log2 domain) -------------
        v8f S[4];
#pragma unroll
        for (int j = 0; j < 4; ++j) {      // 16-key M chunks
            v8f acc = {};
#pragma unroll
            for (int c = 0; c < 2; ++c) {  // K=32 chunks over head dim
                // A layout: lane m=n is the key row; elems 0-7 -> k=8hh+e,
                // elems 8-15 -> k=16+8hh+e  (two contiguous b128 loads).
                const _Float16* p =
                    kb + kbat + (size_t)(key0 + j * 16 + n) * H_ + c * 32 + 8 * hh;
                v16h Af = cat16(*(const v8h*)p, *(const v8h*)(p + 16));
                acc = __builtin_amdgcn_wmma_f32_16x16x32_f16(
                          false, Af, false, Qf[c], (short)0, acc, false, false);
            }
            S[j] = acc;
        }

        // ---- per-lane online softmax over this lane's 32 keys ----------
        float smax = -1e30f;
#pragma unroll
        for (int j = 0; j < 4; ++j)
#pragma unroll
            for (int r = 0; r < 8; ++r) smax = fmaxf(smax, S[j][r]);
        smax = fmaxf(smax, __shfl_xor(smax, 16, 32));   // other 32 keys

        const float mnew  = fmaxf(mrun, smax);
        const float alpha = __builtin_amdgcn_exp2f(mrun - mnew);
        float psum = 0.f;
        // P^T element (key 16j+r+8hh, query n) -> Plds[n][16j+r+8hh]
#pragma unroll
        for (int j = 0; j < 4; ++j)
#pragma unroll
            for (int r = 0; r < 8; ++r) {
                const float p = __builtin_amdgcn_exp2f(S[j][r] - mnew);
                psum += p;
                Plds[wv][n][j * 16 + r + 8 * hh] = (_Float16)p;
            }
        psum += __shfl_xor(psum, 16, 32);
        lrun = lrun * alpha + psum;
        mrun = mnew;
#pragma unroll
        for (int f = 0; f < 4; ++f)
#pragma unroll
            for (int r = 0; r < 8; ++r) O[f][r] *= alpha;   // uniform per lane

        // ---- P^T B-fragments from LDS (compiler inserts ds waits) ------
        v16h Pb[2];
#pragma unroll
        for (int cc = 0; cc < 2; ++cc) {
            const _Float16* pp = &Plds[wv][n][cc * 32 + 8 * hh];
            Pb[cc] = cat16(*(const v8h*)pp, *(const v8h*)(pp + 16));
        }

        // ---- O^T(64x16) += V^T(64x64) * P^T(64x16) ---------------------
#pragma unroll
        for (int f = 0; f < 4; ++f) {      // 16-wide head-dim M chunks
#pragma unroll
            for (int cc = 0; cc < 2; ++cc) {   // K=32 chunks over keys
                const _Float16* p =
                    vt + vbat + (size_t)(f * 16 + n) * T_ + key0 + cc * 32 + 8 * hh;
                v16h Af = cat16(*(const v8h*)p, *(const v8h*)(p + 16));
                O[f] = __builtin_amdgcn_wmma_f32_16x16x32_f16(
                           false, Af, false, Pb[cc], (short)0, O[f], false, false);
            }
        }
    }

    // ---- epilogue: normalize, write column n of the output tile --------
    const float inv = 1.0f / lrun;
    float* po = out + ((size_t)b * T_ + qrow0 + n) * H_ + 8 * hh;
#pragma unroll
    for (int f = 0; f < 4; ++f) {
        float4 lo = { O[f][0] * inv, O[f][1] * inv, O[f][2] * inv, O[f][3] * inv };
        float4 hi = { O[f][4] * inv, O[f][5] * inv, O[f][6] * inv, O[f][7] * inv };
        *(float4*)(po + 16 * f)     = lo;   // head dims 16f+8hh .. +3
        *(float4*)(po + 16 * f + 4) = hi;   // head dims 16f+8hh+4 .. +7
    }
}

// ---------------------------------------------------------------------------
extern "C" void kernel_launch(void* const* d_in, const int* in_sizes, int n_in,
                              void* d_out, int out_size, void* d_ws, size_t ws_size,
                              hipStream_t stream)
{
    (void)in_sizes; (void)n_in; (void)out_size; (void)ws_size;
    const float* x  = (const float*)d_in[0];
    const float* Wk = (const float*)d_in[1];
    const float* Wq = (const float*)d_in[2];
    const float* Wv = (const float*)d_in[3];
    float* out = (float*)d_out;

    const size_t nqkv = (size_t)B_ * T_ * H_;     // 4 MiB f16 each
    _Float16* qb = (_Float16*)d_ws;
    _Float16* kb = qb + nqkv;
    _Float16* vt = kb + nqkv;

    proj_kernel<<<B_ * T_, 64, 0, stream>>>(x, Wk, Wq, Wv, qb, kb, vt);
    attn_kernel<<<B_ * (T_ / 64), 128, 0, stream>>>(qb, kb, vt, out);
}
